// CoordinateLSTMOptimizer_84868553769605
// MI455X (gfx1250) — matrix operand order repaired and checked by reading, not compiled, not run on hardware
//
#include <hip/hip_runtime.h>
#include <hip/hip_bf16.h>

// CDNA5 / gfx1250 coordinate-wise LSTM optimizer.
// 1e6 independent H=20 LSTMs, T=20 steps. Per wave: 16 coordinates.
// gates^T (128x16, padded) = A(W_ext 128x32 f16, constant) x B(h|x|1 32x16 f16)
// via 8x v_wmma_f32_16x16x32_f16 per step. h,c live in registers all 20 steps.
// Activations use native v_tanh_f32 (sigmoid(x) = 0.5 + 0.5*tanh(x/2) -> 1 TRANS op).

typedef __attribute__((ext_vector_type(16))) _Float16 v16h;
typedef __attribute__((ext_vector_type(8)))  float    v8f;

#define LSTM_T 20
#define LSTM_H 20

__device__ __forceinline__ float fexp2f(float x) { return __builtin_amdgcn_exp2f(x); }
__device__ __forceinline__ float frcpf(float x)  { return __builtin_amdgcn_rcpf(x); }

#if defined(__has_builtin)
#if __has_builtin(__builtin_amdgcn_tanhf)
#define LSTM_HAVE_TANH 1
#endif
#endif

__device__ __forceinline__ float tanh_fast(float x) {
#ifdef LSTM_HAVE_TANH
  return __builtin_amdgcn_tanhf(x);   // v_tanh_f32 (gfx1250 TRANS op)
#else
  float e = fexp2f(2.88539008177793f * x);   // exp(2x)
  return 1.0f - 2.0f * frcpf(e + 1.0f);      // (e-1)/(e+1)
#endif
}

__device__ __forceinline__ float sigmf(float x) {
#ifdef LSTM_HAVE_TANH
  // sigmoid(x) = 0.5 + 0.5*tanh(x/2): single v_tanh_f32 instead of exp+add+rcp.
  return fmaf(0.5f, tanh_fast(0.5f * x), 0.5f);
#else
  return frcpf(1.0f + fexp2f(-1.44269504088896f * x));
#endif
}

__device__ __forceinline__ unsigned packh2(_Float16 lo, _Float16 hi) {
  unsigned short l = __builtin_bit_cast(unsigned short, lo);
  unsigned short h = __builtin_bit_cast(unsigned short, hi);
  return (unsigned)l | ((unsigned)h << 16);
}

__global__ __launch_bounds__(256) void
CoordinateLSTMOptimizer_wmma_kernel(const float* __restrict__ grads,
                                    const float* __restrict__ w_ih,
                                    const float* __restrict__ w_hh,
                                    const float* __restrict__ b_ih,
                                    const float* __restrict__ b_hh,
                                    const float* __restrict__ w_fc,
                                    const float* __restrict__ b_fc,
                                    float* __restrict__ out,
                                    int N) {
  const int tid   = threadIdx.x;
  const int lane  = tid & 31;
  const int m     = lane & 15;      // coord within wave group / matrix row id
  const int half  = lane >> 4;      // lane half (K/row-half selector per ISA layouts)
  const bool isH0 = (half == 0);

  const int waveInBlk = tid >> 5;
  const long long waveGlobal = (long long)blockIdx.x * (blockDim.x >> 5) + waveInBlk;
  const long long base = waveGlobal * 16;
  if (base >= N) return;            // uniform exit: keeps EXEC all-ones for WMMA
  const int coord  = (int)base + m;
  const int cclamp = coord < N ? coord : N - 1;

  // ---------------------------------------------------------------
  // A fragments: W_ext, 128 rows x 32 cols, f16, constant across steps.
  //   row r = 32*tg + j  (tg = gate type i/f/g/o, j = hidden unit, j>=20 -> 0)
  //   col k < 20 : w_hh[20*tg + j][k]
  //   col k ==20 : w_ih[20*tg + j]        (B row 20 carries x_t)
  //   col k ==21 : b_ih[..] + b_hh[..]    (B row 21 carries 1.0)
  // A operand, contiguous k order used here (B matches): lane row = m, element i -> k = i + 16*half
  // 32-row-per-gate padding makes all four gate values of hidden unit j land in
  // the SAME lane at the SAME element index across tiles -> zero cross-lane work
  // in the elementwise update.
  // ---------------------------------------------------------------
  v16h A[8];
#pragma unroll
  for (int a = 0; a < 8; ++a) {
    const int r  = 16 * a + m;
    const int tg = r >> 5;
    const int j  = r & 31;
#pragma unroll
    for (int i = 0; i < 16; ++i) {
      const int k = i + 16 * half;
      float val = 0.0f;
      if (j < LSTM_H && k <= 21) {
        const int row = LSTM_H * tg + j;
        if (k < LSTM_H)       val = w_hh[row * LSTM_H + k];
        else if (k == LSTM_H) val = w_ih[row];
        else                  val = b_ih[row] + b_hh[row];
      }
      A[a][i] = (_Float16)val;
    }
  }

  // Per-lane fc weights for the hidden-unit slots this lane owns.
  // half0 owns j = {0..7, 16..19} (slots 0..11); half1 owns j = {8..15} (slots 0..7).
  float wfc[12];
#pragma unroll
  for (int s = 0; s < 12; ++s) {
    int j = isH0 ? (s < 8 ? s : s + 8) : (s < 8 ? s + 8 : -1);
    wfc[s] = (j >= 0) ? w_fc[j] : 0.0f;
  }
  const float bfc = b_fc[0];

  // LSTM state in registers: c as f32 per slot, h packed f16 pairs.
  float cst[12];
#pragma unroll
  for (int s = 0; s < 12; ++s) cst[s] = 0.0f;
  unsigned hp[6] = {0u, 0u, 0u, 0u, 0u, 0u};  // f16 zeros

  const size_t NN = (size_t)N;

  for (int t = 0; t < LSTM_T; ++t) {
    // grads are streamed exactly once -> non-temporal load (TH=NT).
    const float x = __builtin_nontemporal_load(&grads[(size_t)t * NN + (size_t)cclamp]);
    if (t + 1 < LSTM_T)
      __builtin_prefetch(&grads[(size_t)(t + 1) * NN + (size_t)cclamp], 0, 0);

    // Half-swap h packs so each half can assemble its B K-rows.
    unsigned o0 = __shfl_xor(hp[0], 16, 32);
    unsigned o1 = __shfl_xor(hp[1], 16, 32);
    unsigned o2 = __shfl_xor(hp[2], 16, 32);
    unsigned o3 = __shfl_xor(hp[3], 16, 32);
    unsigned o4 = __shfl_xor(hp[4], 16, 32);
    unsigned o5 = __shfl_xor(hp[5], 16, 32);

    // B fragment (32x16 f16): lane column = m; half0 rows K=0..15, half1 K=16..31.
    //   half0: K0..15  = h[0..15]
    //   half1: K16..19 = h[16..19], K20 = x_t, K21 = 1.0, rest 0
    const unsigned xpack = packh2((_Float16)x, (_Float16)1.0f);
    union { unsigned u[8]; v16h v; } B;
    B.u[0] = isH0 ? hp[0] : o4;
    B.u[1] = isH0 ? hp[1] : o5;
    B.u[2] = isH0 ? hp[2] : xpack;
    B.u[3] = isH0 ? hp[3] : 0u;
    B.u[4] = isH0 ? o0 : 0u;
    B.u[5] = isH0 ? o1 : 0u;
    B.u[6] = isH0 ? o2 : 0u;
    B.u[7] = isH0 ? o3 : 0u;

    // 8 WMMAs: full padded gate matrix (128 x 16 coords), bias+input included.
    v8f D[8];
    v8f zacc = {};
#pragma unroll
    for (int a = 0; a < 8; ++a) {
      D[a] = __builtin_amdgcn_wmma_f32_16x16x32_f16(
          false, A[a], false, B.v, (short)0, zacc, false, false);
    }

    // Elementwise LSTM update — zero cross-lane traffic.
    // slot s<8  -> tiles {0,2,4,6}, elem s   (j = s + 8*half)
    // slot s>=8 -> tiles {1,3,5,7}, elem s-8 (j = 16 + s-8, half0 only; half1 sees
    //              padded zero rows -> stays exactly 0, wfc masks the output)
    float acc = 0.0f;
    float hnew[12];
#pragma unroll
    for (int s = 0; s < 12; ++s) {
      const int e  = (s < 8) ? s : s - 8;
      const int a0 = (s < 8) ? 0 : 1;
      const float gi = D[a0 + 0][e];
      const float gf = D[a0 + 2][e];
      const float gg = D[a0 + 4][e];
      const float go = D[a0 + 6][e];
      const float cn = fmaf(sigmf(gf), cst[s], sigmf(gi) * tanh_fast(gg));
      cst[s] = cn;
      const float hn = sigmf(go) * tanh_fast(cn);
      hnew[s] = hn;
      acc = fmaf(hn, wfc[s], acc);
    }

    // Repack h to f16 pairs for the next step's B fragment.
    hp[0] = packh2((_Float16)hnew[0],  (_Float16)hnew[1]);
    hp[1] = packh2((_Float16)hnew[2],  (_Float16)hnew[3]);
    hp[2] = packh2((_Float16)hnew[4],  (_Float16)hnew[5]);
    hp[3] = packh2((_Float16)hnew[6],  (_Float16)hnew[7]);
    hp[4] = packh2((_Float16)hnew[8],  (_Float16)hnew[9]);
    hp[5] = packh2((_Float16)hnew[10], (_Float16)hnew[11]);

    // out_t[coord] = h_new . w_fc + b_fc : reduce across the lane pair.
    const float tot = acc + __shfl_xor(acc, 16, 32) + bfc;
    if (lane < 16 && coord < N)
      __builtin_nontemporal_store(tot, &out[(size_t)t * NN + (size_t)coord]);
  }
}

extern "C" void kernel_launch(void* const* d_in, const int* in_sizes, int n_in,
                              void* d_out, int out_size, void* d_ws, size_t ws_size,
                              hipStream_t stream) {
  const float* grads = (const float*)d_in[0];
  const float* w_ih  = (const float*)d_in[1];
  const float* w_hh  = (const float*)d_in[2];
  const float* b_ih  = (const float*)d_in[3];
  const float* b_hh  = (const float*)d_in[4];
  const float* w_fc  = (const float*)d_in[5];
  const float* b_fc  = (const float*)d_in[6];
  float* out = (float*)d_out;

  const int N = in_sizes[0] / LSTM_T;              // grads is (T, N)
  const int coordsPerBlock = 128;                  // 8 waves x 16 coords
  const int blocks = (N + coordsPerBlock - 1) / coordsPerBlock;

  hipLaunchKernelGGL(CoordinateLSTMOptimizer_wmma_kernel,
                     dim3(blocks), dim3(256), 0, stream,
                     grads, w_ih, w_hh, b_ih, b_hh, w_fc, b_fc, out, N);
}
